// WindowAttention_68925635167008
// MI455X (gfx1250) — matrix operand order repaired and checked by reading, not compiled, not run on hardware
//
#include <hip/hip_runtime.h>
#include <hip/hip_bf16.h>

typedef _Float16 v16h __attribute__((ext_vector_type(16)));
typedef _Float16 v8h  __attribute__((ext_vector_type(8)));
typedef float    v8f  __attribute__((ext_vector_type(8)));
typedef float    v4f  __attribute__((ext_vector_type(4)));

#define NTOK   49      // window tokens (7x7)
#define NPAD   64      // padded to 4 WMMA tiles
#define HD     32      // head dim
#define NH     12      // heads
#define CDIM   384     // channels
#define WAVES  4       // waves per block (one head each)
#define VTSTR  72      // padded LDS row stride (halves): 144B, 16B-aligned runs

__global__ __launch_bounds__(32 * WAVES, 1)
void win_attn_kernel(const float* __restrict__ q,
                     const float* __restrict__ k,
                     const float* __restrict__ v,
                     const float* __restrict__ bias_table,
                     const float* __restrict__ mask,
                     float* __restrict__ out,
                     int nW)
{
    __shared__ __align__(16) _Float16 lds_vt[WAVES * 32 * VTSTR];

    const int lane = threadIdx.x & 31;
    const int wave = threadIdx.x >> 5;
    const int b    = blockIdx.x;
    const int h    = blockIdx.y * WAVES + wave;   // 0..11 exactly
    const int w    = b % nW;                      // mask window index
    const int nn   = lane & 15;                   // fragment column lane
    const int gl   = lane >> 4;                   // lane half-group

    const float scale = 0.1767766952966369f;      // 32^-0.5

    const size_t base = (size_t)b * NTOK * CDIM + (size_t)h * HD;
    const float* qb = q + base;
    const float* kb = k + base;
    const float* vb = v + base;

    _Float16* VT = &lds_vt[wave * 32 * VTSTR];

    // ---- Stage V^T into LDS (lane = head-dim d); packed f16 pair stores ----
    {
        const float* vcol = vb + lane;            // v[b][t][h*32 + lane]
        unsigned* VTrow32 = (unsigned*)(VT + lane * VTSTR);
        #pragma unroll
        for (int t = 0; t < 48; t += 2) {
            _Float16 h0 = (_Float16)vcol[(size_t)t * CDIM];
            _Float16 h1 = (_Float16)vcol[(size_t)(t + 1) * CDIM];
            unsigned p = (unsigned)__builtin_bit_cast(unsigned short, h0) |
                         ((unsigned)__builtin_bit_cast(unsigned short, h1) << 16);
            VTrow32[t >> 1] = p;
        }
        {   // tokens 48 (real) + 49 (pad zero)
            _Float16 h0 = (_Float16)vcol[(size_t)48 * CDIM];
            VTrow32[24] = (unsigned)__builtin_bit_cast(unsigned short, h0);
        }
        #pragma unroll
        for (int t = 50; t < NPAD; t += 2)        // zero pad 50..63
            VTrow32[t >> 1] = 0u;
    }

    // ---- Persistent K A-fragments from global (float4 vector loads) ----
    // A (16x32 f16): lane m=nn; K elems {kb0..kb0+7, kb0+16..kb0+23}, kb0=8*gl
    v16h aK[4];
    #pragma unroll
    for (int jt = 0; jt < 4; ++jt) {
        int tok = jt * 16 + nn; if (tok > NTOK - 1) tok = NTOK - 1;   // clamp pads
        const float* row = kb + (size_t)tok * CDIM + gl * 8;
        v4f a0 = *(const v4f*)(row);
        v4f a1 = *(const v4f*)(row + 4);
        v4f a2 = *(const v4f*)(row + 16);
        v4f a3 = *(const v4f*)(row + 20);
        #pragma unroll
        for (int e = 0; e < 4; ++e) {
            aK[jt][e]      = (_Float16)a0[e];
            aK[jt][e + 4]  = (_Float16)a1[e];
            aK[jt][e + 8]  = (_Float16)a2[e];
            aK[jt][e + 12] = (_Float16)a3[e];
        }
    }

    // ---- Persistent V^T A-fragments from LDS (16B ds_load_b128 runs) ----
    asm volatile("s_wait_dscnt 0" ::: "memory");
    v16h aVT[2][2];
    #pragma unroll
    for (int mt = 0; mt < 2; ++mt) {
        const _Float16* vrow = VT + (mt * 16 + nn) * VTSTR;
        #pragma unroll
        for (int kt = 0; kt < 2; ++kt) {
            const int kb0 = kt * 32 + gl * 16;
            v8h lo = *(const v8h*)(vrow + kb0);
            v8h hi = *(const v8h*)(vrow + kb0 + 8);
            #pragma unroll
            for (int e = 0; e < 8; ++e) {
                aVT[mt][kt][e]     = lo[e];
                aVT[mt][kt][e + 8] = hi[e];
            }
        }
    }

    // ======== Per query-tile pipeline (NOT unrolled: caps VGPR pressure) ====
    #pragma unroll 1
    for (int it = 0; it < 4; ++it) {
        // Q^T B-fragment (32x16 f16): lane n=nn; K elems {8*gl..8*gl+7, 8*gl+16..8*gl+23}
        v16h bQ;
        {
            int tok = it * 16 + nn; if (tok > NTOK - 1) tok = NTOK - 1;
            const float* row = qb + (size_t)tok * CDIM + gl * 8;
            v4f q0 = *(const v4f*)(row);
            v4f q1 = *(const v4f*)(row + 4);
            v4f q2 = *(const v4f*)(row + 16);
            v4f q3 = *(const v4f*)(row + 20);
            #pragma unroll
            for (int e = 0; e < 4; ++e) {
                bQ[e]      = (_Float16)(q0[e] * scale);
                bQ[e + 4]  = (_Float16)(q1[e] * scale);
                bQ[e + 8]  = (_Float16)(q2[e] * scale);
                bQ[e + 12] = (_Float16)(q3[e] * scale);
            }
        }

        // S^T column tiles for this it: S^T[j, i=it*16+nn]
        v8f ST[4];
        #pragma unroll
        for (int jt = 0; jt < 4; ++jt) {
            v8f cz = {};
            ST[jt] = __builtin_amdgcn_wmma_f32_16x16x32_f16(
                false, aK[jt], false, bQ, (short)0, cz, false, false);
        }

        // Bias + mask: branch-free. Clamp addresses, select values (v_cndmask).
        const int i  = it * 16 + nn;              // query index
        const int ic = (i < NTOK) ? i : (NTOK - 1);
        const int ih = ic / 7, iw = ic - 7 * ih;
        const float* mrow = mask + ((size_t)w * NTOK + ic) * NTOK;
        const int ibase = ih * 13 + iw + 84;      // (ih+6)*13 + (iw+6) - j-part
        #pragma unroll
        for (int jt = 0; jt < 4; ++jt) {
            #pragma unroll
            for (int r = 0; r < 8; ++r) {
                const int j  = jt * 16 + 8 * gl + r;              // key index
                const int jc = (j < NTOK) ? j : (NTOK - 1);
                const int jh = jc / 7, jw = jc - 7 * jh;
                const int rel = ibase - (jh * 13 + jw);
                const float add = bias_table[rel * NH + h] + mrow[jc];
                const float s = ST[jt][r] + add;
                ST[jt][r] = (j < NTOK) ? s : -1e30f;              // pad keys
            }
        }

        // Softmax over keys j: registers + one shfl_xor(16); defer 1/sum to O
        float m = -1e30f;
        #pragma unroll
        for (int jt = 0; jt < 4; ++jt)
            #pragma unroll
            for (int r = 0; r < 8; ++r)
                m = fmaxf(m, ST[jt][r]);
        m = fmaxf(m, __shfl_xor(m, 16, 32));

        float sum = 0.f;
        #pragma unroll
        for (int jt = 0; jt < 4; ++jt)
            #pragma unroll
            for (int r = 0; r < 8; ++r) {
                float p = __expf(ST[jt][r] - m);
                ST[jt][r] = p;
                sum += p;
            }
        sum += __shfl_xor(sum, 16, 32);
        const float rinv = __frcp_rn(sum);        // applied to O at the end

        // O^T[:, it-tile] = V^T * P^T ; P^T B-fragments via 2 shuffles/reg
        //   gl=0: e=r -> own ST[2kt][r];        e=r+8 -> partner ST[2kt][r]
        //   gl=1: e=r -> partner ST[2kt+1][r];  e=r+8 -> own ST[2kt+1][r]
        v8f O[2];
        O[0] = (v8f){};
        O[1] = (v8f){};
        #pragma unroll
        for (int kt = 0; kt < 2; ++kt) {
            v16h bP;
            #pragma unroll
            for (int r = 0; r < 8; ++r) {
                float x0  = ST[kt * 2 + 0][r];
                float x1  = ST[kt * 2 + 1][r];
                float x0p = __shfl_xor(x0, 16, 32);
                float x1p = __shfl_xor(x1, 16, 32);
                bP[r]     = (_Float16)((gl == 0) ? x0  : x1p);
                bP[r + 8] = (_Float16)((gl == 0) ? x0p : x1);
            }
            #pragma unroll
            for (int mt = 0; mt < 2; ++mt)
                O[mt] = __builtin_amdgcn_wmma_f32_16x16x32_f16(
                    false, aVT[mt][kt], false, bP, (short)0, O[mt], false, false);
        }

        // Store out[b][i][h*32 + d]: normalize by 1/sum; 2x b128 stores per mt
        if (i < NTOK) {
            float* orow = out + (size_t)b * NTOK * CDIM + (size_t)i * CDIM
                              + h * HD + 8 * gl;
            #pragma unroll
            for (int mt = 0; mt < 2; ++mt) {
                v4f o0 = { O[mt][0] * rinv, O[mt][1] * rinv,
                           O[mt][2] * rinv, O[mt][3] * rinv };
                v4f o1 = { O[mt][4] * rinv, O[mt][5] * rinv,
                           O[mt][6] * rinv, O[mt][7] * rinv };
                *(v4f*)(orow + mt * 16)     = o0;
                *(v4f*)(orow + mt * 16 + 4) = o1;
            }
        }
    }
}

extern "C" void kernel_launch(void* const* d_in, const int* in_sizes, int n_in,
                              void* d_out, int out_size, void* d_ws, size_t ws_size,
                              hipStream_t stream) {
    const float* q    = (const float*)d_in[0];
    const float* k    = (const float*)d_in[1];
    const float* v    = (const float*)d_in[2];
    const float* bias = (const float*)d_in[3];
    const float* mask = (const float*)d_in[4];
    float*       out  = (float*)d_out;

    const int Bw = in_sizes[0] / (NTOK * CDIM);        // 2048 windows
    const int nW = in_sizes[4] / (NTOK * NTOK);        // 64 mask windows

    dim3 grid(Bw, NH / WAVES);                         // (2048, 3)
    dim3 block(32 * WAVES);                            // 4 wave32 = 4 heads
    win_attn_kernel<<<grid, block, 0, stream>>>(q, k, v, bias, mask, out, nW);
}
